// Seq2seq_model_62079457296597
// MI455X (gfx1250) — matrix operand order repaired and checked by reading, not compiled, not run on hardware
//
#include <hip/hip_runtime.h>
#include <hip/hip_bf16.h>

typedef __bf16 bf16_t;
typedef __attribute__((ext_vector_type(16))) __bf16 v16b;
typedef __attribute__((ext_vector_type(8)))  __bf16 v8b;
typedef __attribute__((ext_vector_type(8)))  float  v8f;

#define B_  64
#define T_  256
#define F_  64
#define U_  512
#define NG  2048   /* 4*U */
#define K0  576    /* F_+U_  (layer-0 K) */
#define K1  1024   /* 2*U_   (layer-1 K, also h-row width) */
#define NWG 32
#define NKC0 (K0/32)   /* 18 */
#define NKC1 (K1/32)   /* 32 */
/* LDS: per-WG weight slice in fragment-major layout, 512 bf16 (1KB) per frag */
#define L1_OFF (4 * NKC0 * 512)              /* layer-1 frags after 72 layer-0 frags */
#define SMEM_ELEMS (4 * (NKC0 + NKC1) * 512) /* 102400 bf16 = 200KB */

// ---- branchless fast activations (single v_exp_f32 + v_rcp_f32 each) ----
static __device__ __forceinline__ float fast_sigmoid(float x) {
  return __builtin_amdgcn_rcpf(1.0f + __expf(-x));
}
static __device__ __forceinline__ float fast_tanh(float x) {
  // exp->inf => 1-0 = 1 ; exp->0 => 1-2 = -1  (branchless, endpoint-exact)
  return 1.0f - 2.0f * __builtin_amdgcn_rcpf(1.0f + __expf(2.0f * x));
}

static __device__ __forceinline__ v8f wmma_bf16(v16b a, v16b b, v8f c) {
  return __builtin_amdgcn_wmma_f32_16x16x32_bf16(false, a, false, b, (short)0, c,
                                                 false, false);
}

// A fragment: 16x32 bf16 tile from a row-major matrix (contiguous k = 16h + s layout).
static __device__ __forceinline__ v16b load_a_frag(const bf16_t* base, long row_stride,
                                                   int m_base, int k_base, int lane) {
  int row = m_base + (lane & 15);
  int k0  = k_base + ((lane >> 4) << 4);
  const bf16_t* p = base + (long)row * row_stride + k0;
  union { v16b v; v8b h[2]; } u;
  u.h[0] = *(const v8b*)(p);
  u.h[1] = *(const v8b*)(p + 8);
  return u.v;
}

// B fragment from global W^T [N][K] (used only by the projection kernel).
static __device__ __forceinline__ v16b load_b_frag(const bf16_t* wt, long k_stride,
                                                   int n_base, int k_base, int lane) {
  const bf16_t* p = wt + (long)(n_base + (lane & 15)) * k_stride
                       + k_base + ((lane >> 4) << 4);
  return *(const v16b*)p;
}

// B fragment from LDS, fragment-major: frag = 512 bf16, lane's 16 elems contiguous.
static __device__ __forceinline__ v16b load_b_lds(const bf16_t* frag_base, int lane) {
  const bf16_t* p = frag_base + (lane << 4);
  union { v16b v; v8b h[2]; } u;
  u.h[0] = *(const v8b*)(p);
  u.h[1] = *(const v8b*)(p + 8);
  return u.v;
}

// Copy this WG's weight slice (4 gate chunks x nKc K-chunks) from global W^T
// into LDS in fragment layout. 16B chunks, grid-stride over the workgroup.
static __device__ void stage_weights(const bf16_t* __restrict__ wt, long k_stride,
                                     int nKc, int hcol, bf16_t* __restrict__ dst) {
  const int total = 4 * nKc * 64;            // 16B chunks (64 per 1KB fragment)
  for (int idx = threadIdx.x; idx < total; idx += blockDim.x) {
    const int chunk = idx & 63;
    const int frag  = idx >> 6;
    const int c  = frag / nKc;
    const int kc = frag % nKc;
    const int l    = chunk >> 1;             // lane slot 0..31
    const int half = chunk & 1;              // which 16B half of the 32B slot
    const bf16_t* g = wt + (long)(c * U_ + hcol + (l & 15)) * k_stride
                         + kc * 32 + ((l >> 4) << 4) + half * 8;
    bf16_t* d = dst + (long)frag * 512 + (l << 4) + half * 8;
    *(uint4*)d = *(const uint4*)g;           // global_load_b128 -> ds_store_b128
  }
}

static __device__ __forceinline__ void grid_sync(unsigned* cnt, unsigned target) {
  __threadfence();
  __syncthreads();
  if (threadIdx.x == 0) {
    __hip_atomic_fetch_add(cnt, 1u, __ATOMIC_ACQ_REL, __HIP_MEMORY_SCOPE_AGENT);
    while (__hip_atomic_load(cnt, __ATOMIC_ACQUIRE, __HIP_MEMORY_SCOPE_AGENT) < target) {
      __builtin_amdgcn_s_sleep(2);
    }
  }
  __syncthreads();
  __threadfence();
}

// ---------------- prep kernels ----------------

__global__ void cvt_f32_bf16(const float* __restrict__ s, bf16_t* __restrict__ d, int n) {
  int i = blockIdx.x * blockDim.x + threadIdx.x;
  int stride = gridDim.x * blockDim.x;
  for (; i < n; i += stride) d[i] = (bf16_t)s[i];
}

// d[n*K + k] = bf16(s[k*N + n])   (W [K,N] f32  ->  W^T [N,K] bf16)
__global__ void transpose_cvt(const float* __restrict__ s, bf16_t* __restrict__ d,
                              int K, int N) {
  long total = (long)K * N;
  long i = (long)blockIdx.x * blockDim.x + threadIdx.x;
  long stride = (long)gridDim.x * blockDim.x;
  for (; i < total; i += stride) {
    long n = i / K, k = i % K;
    d[i] = (bf16_t)s[k * (long)N + n];
  }
}

__global__ void init_state(bf16_t* __restrict__ hbuf, unsigned* __restrict__ cnt) {
  int i = blockIdx.x * blockDim.x + threadIdx.x;
  int stride = gridDim.x * blockDim.x;
  const int n = 2 * B_ * K1;
  for (; i < n; i += stride) hbuf[i] = (bf16_t)0.0f;
  if (blockIdx.x == 0 && threadIdx.x == 0) *cnt = 0u;
}

// ---------------- persistent recurrent kernel ----------------
// Grid: 32 WGs x 128 threads (4 waves). Wave w owns rows [16w,16w+16).
// WG g owns h-columns [16g,16g+16) and the matching 4 gate slices, whose
// weights (200KB) live in LDS for the whole phase.
__global__ __launch_bounds__(128, 1) void lstm_seq_kernel(
    const bf16_t* __restrict__ xb, const bf16_t* __restrict__ tb,
    const bf16_t* __restrict__ eW0t, const float* __restrict__ eb0,
    const bf16_t* __restrict__ eW1t, const float* __restrict__ eb1,
    const bf16_t* __restrict__ dW0t, const float* __restrict__ db0,
    const bf16_t* __restrict__ dW1t, const float* __restrict__ db1,
    bf16_t* __restrict__ hbuf, bf16_t* __restrict__ dech,
    unsigned* __restrict__ cnt)
{
  extern __shared__ bf16_t smem[];           // SMEM_ELEMS bf16 (200KB)
  const int wg   = blockIdx.x;               // 0..31
  const int wave = threadIdx.x >> 5;         // 0..3
  const int lane = threadIdx.x & 31;
  const int m_base = wave << 4;
  const int hcol   = wg << 4;
  const int ncol   = lane & 15;
  const int mrow   = ((lane >> 4) << 3);
  const bf16_t* L0 = smem;
  const bf16_t* L1 = smem + L1_OFF;

  float c0[8], c1[8];
#pragma unroll
  for (int r = 0; r < 8; ++r) { c0[r] = 0.0f; c1[r] = 0.0f; }

  unsigned sync_no = 0;

  for (int ph = 0; ph < 2; ++ph) {           // 0 = encoder, 1 = decoder
    const bf16_t* X  = ph ? tb   : xb;
    const float* Bb0 = ph ? db0 : eb0;
    const float* Bb1 = ph ? db1 : eb1;

    // Stage this phase's weight slices into LDS (reused for all 256 steps).
    stage_weights(ph ? dW0t : eW0t, K0, NKC0, hcol, smem);
    stage_weights(ph ? dW1t : eW1t, K1, NKC1, hcol, smem + L1_OFF);
    __syncthreads();

    const float bi0 = Bb0[0 * U_ + hcol + ncol];
    const float bj0 = Bb0[1 * U_ + hcol + ncol];
    const float bf0 = Bb0[2 * U_ + hcol + ncol];
    const float bo0 = Bb0[3 * U_ + hcol + ncol];
    const float bi1 = Bb1[0 * U_ + hcol + ncol];
    const float bj1 = Bb1[1 * U_ + hcol + ncol];
    const float bf1 = Bb1[2 * U_ + hcol + ncol];
    const float bo1 = Bb1[3 * U_ + hcol + ncol];

    for (int t = 0; t < T_; ++t) {
      const int s = (ph << 8) | t;
      bf16_t* hcur  = hbuf + (size_t)(s & 1) * (B_ * K1);
      bf16_t* hprev = hbuf + (size_t)((s & 1) ^ 1) * (B_ * K1);

      // -------- layer 0: gates = [x_t | h0(t-1)] @ W0 --------
      {
        v8f ai = {}, aj = {}, af = {}, ao = {};
#pragma unroll 2
        for (int kc = 0; kc < NKC0; ++kc) {
          v16b a = (kc < 2)
            ? load_a_frag(X + (long)t * F_, (long)T_ * F_, m_base, kc * 32, lane)
            : load_a_frag(hprev, K1, m_base, kc * 32 - F_, lane);
          ai = wmma_bf16(a, load_b_lds(L0 + ((0 * NKC0 + kc) << 9), lane), ai);
          aj = wmma_bf16(a, load_b_lds(L0 + ((1 * NKC0 + kc) << 9), lane), aj);
          af = wmma_bf16(a, load_b_lds(L0 + ((2 * NKC0 + kc) << 9), lane), af);
          ao = wmma_bf16(a, load_b_lds(L0 + ((3 * NKC0 + kc) << 9), lane), ao);
        }
        if (t + 1 < T_)  // prefetch next step's x rows (global_prefetch_b8)
          __builtin_prefetch(X + (long)(m_base + ncol) * T_ * F_ + (long)(t + 1) * F_, 0, 1);
#pragma unroll
        for (int r = 0; r < 8; ++r) {
          const int m = m_base + r + mrow;
          const float ig = fast_sigmoid(ai[r] + bi0);
          const float jg = fast_tanh(aj[r] + bj0);
          const float fg = fast_sigmoid(af[r] + bf0 + 1.0f);   // FORGET_BIAS
          const float og = fast_sigmoid(ao[r] + bo0);
          c0[r] = fg * c0[r] + ig * jg;
          const float hv = og * fast_tanh(c0[r]);
          hcur[(long)m * K1 + hcol + ncol] = (bf16_t)hv;
        }
      }
      grid_sync(cnt, (++sync_no) * NWG);

      // -------- layer 1: gates = [h0(t) | h1(t-1)] @ W1 --------
      {
        v8f zi = {}, zj = {}, zf = {}, zo = {};
#pragma unroll 2
        for (int kc = 0; kc < NKC1; ++kc) {
          const bf16_t* hb = (kc < 16) ? hcur : hprev;  // h0(t) | h1(t-1)
          v16b a = load_a_frag(hb, K1, m_base, kc * 32, lane);
          zi = wmma_bf16(a, load_b_lds(L1 + ((0 * NKC1 + kc) << 9), lane), zi);
          zj = wmma_bf16(a, load_b_lds(L1 + ((1 * NKC1 + kc) << 9), lane), zj);
          zf = wmma_bf16(a, load_b_lds(L1 + ((2 * NKC1 + kc) << 9), lane), zf);
          zo = wmma_bf16(a, load_b_lds(L1 + ((3 * NKC1 + kc) << 9), lane), zo);
        }
#pragma unroll
        for (int r = 0; r < 8; ++r) {
          const int m = m_base + r + mrow;
          const float ig = fast_sigmoid(zi[r] + bi1);
          const float jg = fast_tanh(zj[r] + bj1);
          const float fg = fast_sigmoid(zf[r] + bf1 + 1.0f);
          const float og = fast_sigmoid(zo[r] + bo1);
          c1[r] = fg * c1[r] + ig * jg;
          const float hv = og * fast_tanh(c1[r]);
          const bf16_t hv16 = (bf16_t)hv;
          hcur[(long)m * K1 + U_ + hcol + ncol] = hv16;
          if (ph == 1) dech[((long)t * B_ + m) * U_ + hcol + ncol] = hv16;
        }
      }
      grid_sync(cnt, (++sync_no) * NWG);
    }
  }
}

// ---------------- output projection + length mask ----------------
// dech [T*B, U] bf16 @ out_W^T [F][U] bf16  ->  d_out [B,T,F] f32 (masked)
__global__ __launch_bounds__(256, 2) void proj_kernel(
    const bf16_t* __restrict__ dech, const bf16_t* __restrict__ oWt,
    const float* __restrict__ ob, const int* __restrict__ seqlen,
    float* __restrict__ out)
{
  const int wave = threadIdx.x >> 5;
  const int lane = threadIdx.x & 31;
  const int rowTile = blockIdx.x * 8 + wave;   // 0..1023 (16384 rows / 16)
  const int m_base  = rowTile << 4;
  const int n_base  = blockIdx.y << 4;         // 0..48
  v8f acc = {};
#pragma unroll 4
  for (int kc = 0; kc < U_ / 32; ++kc) {
    v16b a = load_a_frag(dech, U_, m_base, kc * 32, lane);
    v16b b = load_b_frag(oWt, U_, n_base, kc * 32, lane);
    acc = wmma_bf16(a, b, acc);
  }
  const int f = n_base + (lane & 15);
  const float bias = ob[f];
#pragma unroll
  for (int r = 0; r < 8; ++r) {
    const int m = m_base + r + ((lane >> 4) << 3);  // m = t*64 + b
    const int t = m >> 6;
    const int bidx = m & 63;
    const float v = (t < seqlen[bidx]) ? (acc[r] + bias) : 0.0f;
    out[((long)bidx * T_ + t) * F_ + f] = v;
  }
}

// ---------------- host launcher ----------------

extern "C" void kernel_launch(void* const* d_in, const int* in_sizes, int n_in,
                              void* d_out, int out_size, void* d_ws, size_t ws_size,
                              hipStream_t stream) {
  const float* inputs  = (const float*)d_in[0];
  const float* targets = (const float*)d_in[1];
  const int*   seqlen  = (const int*)d_in[2];
  const float* eW0 = (const float*)d_in[3];
  const float* eb0 = (const float*)d_in[4];
  const float* eW1 = (const float*)d_in[5];
  const float* eb1 = (const float*)d_in[6];
  const float* dW0 = (const float*)d_in[7];
  const float* db0 = (const float*)d_in[8];
  const float* dW1 = (const float*)d_in[9];
  const float* db1 = (const float*)d_in[10];
  const float* oW  = (const float*)d_in[11];
  const float* ob  = (const float*)d_in[12];

  size_t off = 0;
  auto take = [&](size_t bytes) -> void* {
    void* p = (char*)d_ws + off;
    off += (bytes + 255) & ~(size_t)255;
    return p;
  };
  bf16_t* xb   = (bf16_t*)take((size_t)B_ * T_ * F_ * 2);
  bf16_t* tb   = (bf16_t*)take((size_t)B_ * T_ * F_ * 2);
  bf16_t* eW0t = (bf16_t*)take((size_t)NG * K0 * 2);
  bf16_t* eW1t = (bf16_t*)take((size_t)NG * K1 * 2);
  bf16_t* dW0t = (bf16_t*)take((size_t)NG * K0 * 2);
  bf16_t* dW1t = (bf16_t*)take((size_t)NG * K1 * 2);
  bf16_t* oWt  = (bf16_t*)take((size_t)F_ * U_ * 2);
  bf16_t* hbuf = (bf16_t*)take((size_t)2 * B_ * K1 * 2);
  bf16_t* dech = (bf16_t*)take((size_t)T_ * B_ * U_ * 2);
  unsigned* cnt = (unsigned*)take(256);

  cvt_f32_bf16<<<128, 256, 0, stream>>>(inputs,  xb, B_ * T_ * F_);
  cvt_f32_bf16<<<128, 256, 0, stream>>>(targets, tb, B_ * T_ * F_);
  transpose_cvt<<<256, 256, 0, stream>>>(eW0, eW0t, K0, NG);
  transpose_cvt<<<256, 256, 0, stream>>>(eW1, eW1t, K1, NG);
  transpose_cvt<<<256, 256, 0, stream>>>(dW0, dW0t, K0, NG);
  transpose_cvt<<<256, 256, 0, stream>>>(dW1, dW1t, K1, NG);
  transpose_cvt<<<64, 256, 0, stream>>>(oW, oWt, U_, F_);
  init_state<<<64, 256, 0, stream>>>(hbuf, cnt);

  lstm_seq_kernel<<<NWG, 128, (int)(SMEM_ELEMS * sizeof(bf16_t)), stream>>>(
      xb, tb, eW0t, eb0, eW1t, eb1, dW0t, db0, dW1t, db1, hbuf, dech, cnt);

  proj_kernel<<<dim3(128, 4), 256, 0, stream>>>(dech, oWt, ob, seqlen, (float*)d_out);
}
